// S5SSM_8615704395883
// MI455X (gfx1250) — matrix operand order repaired and checked by reading, not compiled, not run on hardware
//
#include <hip/hip_runtime.h>

typedef __bf16 v16bf __attribute__((ext_vector_type(16)));
typedef float  v8f   __attribute__((ext_vector_type(8)));

#define WMMA_BF16(A,B,C) __builtin_amdgcn_wmma_f32_16x16x32_bf16(false,(A),false,(B),(short)0,(C),false,false)

constexpr int Lx = 131072;        // sequence length
constexpr int Hx = 128;           // channels
constexpr int Px = 128;           // state dim
constexpr int Tc = 512;           // rows per chunk
constexpr int NCHUNK = Lx / Tc;   // 256 chunks
constexpr int SLAB = 16;          // rows per slab (one M tile)
constexpr int NSLAB = Tc / SLAB;  // 32 slabs per chunk

constexpr int BU_STR = 260;       // f32 stride, Bu slab (256 cols + pad)
constexpr int XH_STR = 136;       // bf16 stride, x slab (128 cols + pad)
constexpr int HS_STR = 264;       // bf16 stride, hs slab (256 cols + pad)

// dynamic-LDS layout (bytes)
constexpr int OFF_BU  = 0;                         // f32 [SLAB*BU_STR] 16640
constexpr int OFF_XH  = 16640;                     // bf16[SLAB*XH_STR] 4352
constexpr int OFF_XL  = OFF_XH + 4352;             // 20992
constexpr int OFF_HSH = OFF_XL + 4352;             // 25344 (FINAL only)
constexpr int OFF_HSL = OFF_HSH + 8448;            // 33792
constexpr int OFF_DL  = OFF_HSL + 8448;            // 42240
constexpr int LDS_PASS1 = OFF_HSH;                 // 25344 B
constexpr int LDS_PASS3 = OFF_DL + 512;            // 42752 B

// ---------------------------------------------------------------------------
// helpers
// ---------------------------------------------------------------------------
// A-fragment swizzle: place column K of a row so that the 16 bf16 a lane needs
// for WMMA fragment (kb, half) are contiguous:
//   frag element e <-> K = kb*32 + 8*half + e + (e>=8 ? 8 : 0)
//   slot(K) = kb*32 + half*16 + (K&7) + 8*((K>>4)&1),  half = (K>>3)&1
__device__ inline int fragslot(int K) {
  int kb = K >> 5, r = K & 31;
  int half = (r >> 3) & 1;
  int idx = (r & 7) + ((r >> 4) << 3);
  return kb * 32 + half * 16 + idx;
}

__device__ inline v16bf lds_frag(const __bf16* p) {
  v16bf r;
  __builtin_memcpy(&r, p, sizeof(v16bf));  // 2x ds_load_b128 (16B aligned)
  return r;
}

__device__ inline void store_split(__bf16* hi, __bf16* lo, int idx, float v) {
  __bf16 h = (__bf16)v;
  hi[idx] = h;
  lo[idx] = (__bf16)(v - (float)h);
}

// ---------------------------------------------------------------------------
// Kernel 0: discretization + pre-swizzled bf16 hi/lo B and C matrices.
// Bmat: (2P columns x H k), column-major-K:  Bmat[n*H + k]
//   n <  P : dB_re[n][k] ;  n >= P : dB_im[n-P][k]
// Cmat: (H columns x 2P k), column-major-K:  Cmat[n*2P + k]
//   k <  P :  2*C_re[n][k] ; k >= P : -2*C_im[n][k-P]   (folds CONJ_SYM 2x)
// Also dA, dA^512, dA^(512*64) for the carry scan.
// ---------------------------------------------------------------------------
__global__ void setup_kernel(const float* __restrict__ Arl,
                             const float* __restrict__ Aim,
                             const float* __restrict__ Bre,
                             const float* __restrict__ Bim,
                             const float* __restrict__ Cre,
                             const float* __restrict__ Cim,
                             const float* __restrict__ invdt,
                             float* __restrict__ wdA, float* __restrict__ wdAT,
                             float* __restrict__ wdATG,
                             __bf16* __restrict__ Bmh, __bf16* __restrict__ Bml,
                             __bf16* __restrict__ Cmh, __bf16* __restrict__ Cml) {
  const int p = threadIdx.x;  // 128 threads
  float idt = invdt[p];
  float dt = (idt > 15.f) ? idt : log1pf(__expf(idt));  // softplus
  float ar = Arl[p], ai = Aim[p];
  float zr = 0.5f * dt * ar, zi = 0.5f * dt * ai;       // 0.5*dt*A
  float dr = 1.f - zr, di = -zi;                        // 1 - z
  float den = dr * dr + di * di;
  float blr = dr / den, bli = -di / den;                // BL = 1/(1-z)
  float or_ = 1.f + zr, oi = zi;                        // 1 + z
  float dAr = blr * or_ - bli * oi;                     // dA = BL*(1+z)
  float dAi = blr * oi + bli * or_;
  wdA[p] = dAr; wdA[Px + p] = dAi;
  float tr = dAr, ti = dAi;  // dA^512 (9 squarings), then ^64 more
#pragma unroll
  for (int k = 0; k < 9; ++k) { float nr = tr*tr - ti*ti, ni = 2.f*tr*ti; tr = nr; ti = ni; }
  wdAT[p] = tr; wdAT[Px + p] = ti;
#pragma unroll
  for (int k = 0; k < 6; ++k) { float nr = tr*tr - ti*ti, ni = 2.f*tr*ti; tr = nr; ti = ni; }
  wdATG[p] = tr; wdATG[Px + p] = ti;

  float sr = blr * dt, si = bli * dt;                   // BL*dt
  for (int h = 0; h < Hx; ++h) {
    float br = Bre[p * Hx + h], bi = Bim[p * Hx + h];
    store_split(Bmh, Bml, p * Hx + h, sr * br - si * bi);         // dB_re
    store_split(Bmh, Bml, (Px + p) * Hx + h, sr * bi + si * br);  // dB_im
  }
  for (int k = 0; k < Px; ++k) {  // thread p owns output column n = p
    store_split(Cmh, Cml, p * (2 * Px) + k, 2.f * Cre[p * Px + k]);
    store_split(Cmh, Cml, p * (2 * Px) + Px + k, -2.f * Cim[p * Px + k]);
  }
}

// ---------------------------------------------------------------------------
// Kernel 2: exclusive scan of chunk carries (single block, 4 groups x 128).
// P_c = dA^512 * P_{c-1} + S_{c-1}, P_0 = 0.  Seed-rescan over 64 chunks/group.
// ---------------------------------------------------------------------------
__global__ void carry_kernel(const float* __restrict__ wS,
                             float* __restrict__ wP,
                             const float* __restrict__ wdAT,
                             const float* __restrict__ wdATG) {
  __shared__ float G[4][2][Px];
  __shared__ float E[4][2][Px];
  const int g = threadIdx.x >> 7;
  const int p = threadIdx.x & 127;
  const int base = g * (NCHUNK / 4);
  const float atr = wdAT[p], ati = wdAT[Px + p];    // dA^512
  const float gtr = wdATG[p], gti = wdATG[Px + p];  // dA^(512*64)
  float hr = 0.f, hi = 0.f;
  for (int j = 0; j < NCHUNK / 4; ++j) {
    int c = base + j;
    float sr = wS[c * 256 + p], si = wS[c * 256 + Px + p];
    float nr = atr * hr - ati * hi + sr;
    float ni = atr * hi + ati * hr + si;
    hr = nr; hi = ni;
  }
  G[g][0][p] = hr; G[g][1][p] = hi;
  __syncthreads();
  if (threadIdx.x < Px) {
    float er = 0.f, ei = 0.f;
#pragma unroll
    for (int gg = 0; gg < 4; ++gg) {
      E[gg][0][p] = er; E[gg][1][p] = ei;
      float cr = G[gg][0][p], ci = G[gg][1][p];
      float nr = gtr * er - gti * ei + cr;
      float ni = gtr * ei + gti * er + ci;
      er = nr; ei = ni;
    }
  }
  __syncthreads();
  hr = E[g][0][p]; hi = E[g][1][p];
  for (int j = 0; j < NCHUNK / 4; ++j) {
    int c = base + j;
    wP[c * 256 + p] = hr; wP[c * 256 + Px + p] = hi;
    float sr = wS[c * 256 + p], si = wS[c * 256 + Px + p];
    float nr = atr * hr - ati * hi + sr;
    float ni = atr * hi + ati * hr + si;
    hr = nr; hi = ni;
  }
}

// ---------------------------------------------------------------------------
// Main chunk kernel. FINAL=false: GEMM1 -> scan -> chunk carry.
//                    FINAL=true : seed from prefix, + GEMM2 -> y.
// One block (8 waves) per chunk; 16-row slabs streamed through LDS. All bf16
// splits are produced once (staging / scan threads), GEMM loops are pure
// ds_load + global_load + v_wmma.
// ---------------------------------------------------------------------------
template <bool FINAL>
__global__ __launch_bounds__(256) void pass_kernel(
    const float* __restrict__ x, const float* __restrict__ Dvec,
    float* __restrict__ y, const float* __restrict__ wdA,
    const __bf16* __restrict__ Bmh, const __bf16* __restrict__ Bml,
    const __bf16* __restrict__ Cmh, const __bf16* __restrict__ Cml,
    float* __restrict__ wS, const float* __restrict__ wP) {
  extern __shared__ char smem[];
  float*  bu  = (float*)(smem + OFF_BU);     // GEMM1 out (f32), scan input
  __bf16* xh  = (__bf16*)(smem + OFF_XH);    // x slab, bf16 hi (frag order)
  __bf16* xl  = (__bf16*)(smem + OFF_XL);    // x slab, bf16 lo
  __bf16* hsh = (__bf16*)(smem + OFF_HSH);   // hs slab, bf16 hi (FINAL)
  __bf16* hsl = (__bf16*)(smem + OFF_HSL);   // hs slab, bf16 lo (FINAL)
  float*  Dl  = (float*)(smem + OFF_DL);

  const int c = blockIdx.x;
  const int tid = threadIdx.x;
  const int w = tid >> 5;      // wave 0..7
  const int lane = tid & 31;
  const int half = lane >> 4;
  const int m = lane & 15;     // A row / B,C,D column within tile

  float hr = 0.f, hi = 0.f, dAr = 0.f, dAi = 0.f;
  int sre = 0, sim = 0;
  if (tid < Px) {
    dAr = wdA[tid]; dAi = wdA[Px + tid];
    sre = tid; sim = Px + tid;
    if (FINAL) { hr = wP[c * 256 + tid]; hi = wP[c * 256 + Px + tid]; }
  }
  if (FINAL && tid < Hx) Dl[tid] = Dvec[tid];

  const int l0c = c * Tc;
  for (int s = 0; s < NSLAB; ++s) {
    const int l0 = l0c + s * SLAB;
    // ---- stage x slab: f32 global -> bf16 hi/lo LDS in fragment order ----
    for (int idx = tid; idx < SLAB * Hx / 4; idx += 256) {
      int r = idx >> 5;               // 32 float4 per row
      int c0 = (idx & 31) << 2;       // col of first element
      const float4 v = *(const float4*)(x + (l0 + r) * Hx + c0);
      int s0 = c0;                    // slots are consecutive for c0..c0+3
      float vv[4] = {v.x, v.y, v.z, v.w};
#pragma unroll
      for (int j = 0; j < 4; ++j) {
        __bf16 h = (__bf16)vv[j];
        xh[r * XH_STR + s0 + j] = h;
        xl[r * XH_STR + s0 + j] = (__bf16)(vv[j] - (float)h);
      }
    }
    __syncthreads();
    // ---- GEMM1: Bu(16x256) = x_slab(16x128) @ [dB_re^T | dB_im^T] --------
    {
      v8f acc0 = {0.f, 0.f, 0.f, 0.f, 0.f, 0.f, 0.f, 0.f};
      v8f acc1 = {0.f, 0.f, 0.f, 0.f, 0.f, 0.f, 0.f, 0.f};
      const int nt0 = 2 * w, nt1 = 2 * w + 1;  // 2 N-tiles of 16 per wave
#pragma unroll
      for (int kb = 0; kb < Hx / 32; ++kb) {
        const int off = kb * 32 + 16 * half;
        const v16bf ah = lds_frag(xh + m * XH_STR + off);
        const v16bf al = lds_frag(xl + m * XH_STR + off);
        const v16bf b0h = *(const v16bf*)(Bmh + (nt0 * 16 + m) * Hx + off);
        const v16bf b0l = *(const v16bf*)(Bml + (nt0 * 16 + m) * Hx + off);
        const v16bf b1h = *(const v16bf*)(Bmh + (nt1 * 16 + m) * Hx + off);
        const v16bf b1l = *(const v16bf*)(Bml + (nt1 * 16 + m) * Hx + off);
        acc0 = WMMA_BF16(ah, b0h, acc0);
        acc0 = WMMA_BF16(al, b0h, acc0);
        acc0 = WMMA_BF16(ah, b0l, acc0);
        acc1 = WMMA_BF16(ah, b1h, acc1);
        acc1 = WMMA_BF16(al, b1h, acc1);
        acc1 = WMMA_BF16(ah, b1l, acc1);
      }
      // C/D layout: VGPR i -> row M = i + 8*half, col N = lane&15
#pragma unroll
      for (int i = 0; i < 8; ++i) {
        bu[(i + 8 * half) * BU_STR + nt0 * 16 + m] = acc0[i];
        bu[(i + 8 * half) * BU_STR + nt1 * 16 + m] = acc1[i];
      }
    }
    __syncthreads();
    // ---- serial scan: h = dA*h + Bu; emit hs as bf16 hi/lo (FINAL) -------
    if (tid < Px) {
#pragma unroll
      for (int i = 0; i < SLAB; ++i) {
        float br = bu[i * BU_STR + tid];
        float bi2 = bu[i * BU_STR + Px + tid];
        float nr = fmaf(dAr, hr, fmaf(-dAi, hi, br));
        float ni = fmaf(dAr, hi, fmaf(dAi, hr, bi2));
        hr = nr; hi = ni;
        if (FINAL) {
          __bf16 h1 = (__bf16)hr;
          hsh[i * HS_STR + sre] = h1;
          hsl[i * HS_STR + sre] = (__bf16)(hr - (float)h1);
          __bf16 h2 = (__bf16)hi;
          hsh[i * HS_STR + sim] = h2;
          hsl[i * HS_STR + sim] = (__bf16)(hi - (float)h2);
        }
      }
    }
    __syncthreads();
    if (FINAL) {
      // ---- GEMM2: y(16x128) = [hs_re|hs_im](16x256) @ Cmat(256x128) ------
      v8f acc = {0.f, 0.f, 0.f, 0.f, 0.f, 0.f, 0.f, 0.f};
#pragma unroll
      for (int kb = 0; kb < (2 * Px) / 32; ++kb) {
        const int off = kb * 32 + 16 * half;
        const v16bf ah = lds_frag(hsh + m * HS_STR + off);
        const v16bf al = lds_frag(hsl + m * HS_STR + off);
        const v16bf bh = *(const v16bf*)(Cmh + (w * 16 + m) * (2 * Px) + off);
        const v16bf bl = *(const v16bf*)(Cml + (w * 16 + m) * (2 * Px) + off);
        acc = WMMA_BF16(ah, bh, acc);
        acc = WMMA_BF16(al, bh, acc);
        acc = WMMA_BF16(ah, bl, acc);
      }
      const int col = w * 16 + m;
      const int sc = col;
#pragma unroll
      for (int i = 0; i < 8; ++i) {
        int M = i + 8 * half;
        float xv = (float)xh[M * XH_STR + sc] + (float)xl[M * XH_STR + sc];
        y[(l0 + M) * Hx + col] = acc[i] + xv * Dl[col];
      }
      __syncthreads();  // protect xh/xl/hsh/hsl before next slab reuse
    }
  }
  if (!FINAL && tid < Px) {
    wS[c * 256 + tid] = hr;
    wS[c * 256 + Px + tid] = hi;
  }
}

// ---------------------------------------------------------------------------
// launch
// ---------------------------------------------------------------------------
extern "C" void kernel_launch(void* const* d_in, const int* in_sizes, int n_in,
                              void* d_out, int out_size, void* d_ws,
                              size_t ws_size, hipStream_t stream) {
  const float* x   = (const float*)d_in[0];
  const float* Arl = (const float*)d_in[1];
  const float* Aim = (const float*)d_in[2];
  const float* Bre = (const float*)d_in[3];
  const float* Bim = (const float*)d_in[4];
  const float* Cre = (const float*)d_in[5];
  const float* Cim = (const float*)d_in[6];
  const float* Dv  = (const float*)d_in[7];
  const float* idt = (const float*)d_in[8];
  float* y = (float*)d_out;

  char* ws = (char*)d_ws;
  float*  wdA   = (float*)(ws);                       // 1 KB
  float*  wdAT  = (float*)(ws + 1024);                // 1 KB
  float*  wdATG = (float*)(ws + 2048);                // 1 KB
  __bf16* Bmh   = (__bf16*)(ws + 4096);               // 64 KB
  __bf16* Bml   = (__bf16*)(ws + 4096 + 65536);       // 64 KB
  __bf16* Cmh   = (__bf16*)(ws + 4096 + 2 * 65536);   // 64 KB
  __bf16* Cml   = (__bf16*)(ws + 4096 + 3 * 65536);   // 64 KB
  float*  wS    = (float*)(ws + 4096 + 4 * 65536);            // 256 KB
  float*  wP    = (float*)(ws + 4096 + 4 * 65536 + 262144);   // 256 KB

  setup_kernel<<<1, 128, 0, stream>>>(Arl, Aim, Bre, Bim, Cre, Cim, idt, wdA,
                                      wdAT, wdATG, Bmh, Bml, Cmh, Cml);
  pass_kernel<false><<<NCHUNK, 256, LDS_PASS1, stream>>>(
      x, nullptr, nullptr, wdA, Bmh, Bml, Cmh, Cml, wS, nullptr);
  carry_kernel<<<1, 512, 0, stream>>>(wS, wP, wdAT, wdATG);
  pass_kernel<true><<<NCHUNK, 256, LDS_PASS3, stream>>>(
      x, Dv, y, wdA, Bmh, Bml, Cmh, Cml, wS, wP);
}
